// InteractMappingNetwork_66563403153902
// MI455X (gfx1250) — matrix operand order repaired and checked by reading, not compiled, not run on hardware
//
#include <hip/hip_runtime.h>
#include <hip/hip_bf16.h>

// ---------------------------------------------------------------------------
// Bidirectional cross-attention (GLIP BiAttention) for MI455X / gfx1250.
// GEMMs via v_wmma_f32_16x16x32_f16; large projections staged through LDS by
// the Tensor Data Mover (tensor_load_to_lds + s_wait_tensorcnt), double-
// buffered with K-step 64, TDM LDS padding for bank-conflict-free ds_load_b128.
// ---------------------------------------------------------------------------

typedef __attribute__((ext_vector_type(16))) _Float16 v16h;
typedef __attribute__((ext_vector_type(8)))  _Float16 v8h;
typedef __attribute__((ext_vector_type(8)))  float    v8f;
typedef __attribute__((ext_vector_type(4)))  unsigned u32x4;
typedef __attribute__((ext_vector_type(8)))  int      i32x8;
typedef __attribute__((ext_vector_type(4)))  int      i32x4;

#if defined(__has_builtin)
#if __has_builtin(__builtin_amdgcn_tensor_load_to_lds) && __has_builtin(__builtin_amdgcn_s_wait_tensorcnt)
#define USE_TDM 1
#endif
#endif

namespace {
constexpr int B  = 4;
constexpr int TV = 4096;
constexpr int TL = 256;
constexpr int VD = 1024;
constexpr int LD = 768;
constexpr int E  = 1024;
constexpr int H  = 16;
constexpr int Dh = 64;            // head dim
constexpr float SCALE = 0.125f;   // Dh^-0.5
constexpr float EPS   = 1e-5f;
constexpr float CLAMP = 50000.0f;
// Staged-GEMM geometry: K-step 64 f16 per stage; each 128B row padded by the
// TDM with 16B -> 144B stride (36 banks; 16 rows hit 16 disjoint bank groups).
constexpr int KSTEP = 64;
constexpr int LROW  = 72;              // halves per LDS row (64 + 8 pad)
constexpr int A_HALVES = 128 * LROW;   // 9216
constexpr int B_HALVES = 64 * LROW;    // 4608
}

// ---- order-preserving float<->uint encoding for global atomic max ----------
__device__ __forceinline__ unsigned enc_f32(float f) {
  unsigned u = __float_as_uint(f);
  return (u & 0x80000000u) ? ~u : (u | 0x80000000u);
}
__device__ __forceinline__ float dec_f32(unsigned e) {
  unsigned u = (e & 0x80000000u) ? (e & 0x7FFFFFFFu) : ~e;
  return __uint_as_float(u);
}

// ---- WMMA fragment loaders (contiguous k order, same for A and B) ----------------------------
// A (16x32 f16): lane m=lane&15, 16 contiguous K halves at K=(lane>>4)*16.
__device__ __forceinline__ v16h ldA(const _Float16* p) {
  v8h lo = *(const v8h*)(p);
  v8h hi = *(const v8h*)(p + 8);
  v16h r;
#pragma unroll
  for (int e = 0; e < 8; ++e) { r[e] = lo[e]; r[e + 8] = hi[e]; }
  return r;
}
// B (32x16 f16): lane n=lane&15, 16 contiguous K halves at K=(lane>>4)*16.
__device__ __forceinline__ v16h ldB(const _Float16* p) {
  v8h lo = *(const v8h*)(p);
  v8h hi = *(const v8h*)(p + 8);
  v16h r;
#pragma unroll
  for (int e = 0; e < 8; ++e) { r[e] = lo[e]; r[e + 8] = hi[e]; }
  return r;
}

__device__ __forceinline__ void wmma4(const v16h& a0, const v16h& a1,
                                      const v16h& b0, const v16h& b1, v8f c[2][2]) {
  c[0][0] = __builtin_amdgcn_wmma_f32_16x16x32_f16(false, a0, false, b0, (short)0, c[0][0], false, false);
  c[0][1] = __builtin_amdgcn_wmma_f32_16x16x32_f16(false, a0, false, b1, (short)0, c[0][1], false, false);
  c[1][0] = __builtin_amdgcn_wmma_f32_16x16x32_f16(false, a1, false, b0, (short)0, c[1][0], false, false);
  c[1][1] = __builtin_amdgcn_wmma_f32_16x16x32_f16(false, a1, false, b1, (short)0, c[1][1], false, false);
}

// 32x32 wave tile from global memory (B given as rows of [N,K] matrix).
__device__ __forceinline__ void mm32x32(const _Float16* __restrict__ A, int lda,
                                        const _Float16* __restrict__ Bw, int ldb,
                                        int K, v8f c[2][2]) {
  const int lane = threadIdx.x & 31;
  const int lh = lane & 15;
  const int lq = lane >> 4;
  for (int k = 0; k < K; k += 32) {
    v16h a0 = ldA(A + (size_t)lh        * lda + k + lq * 16);
    v16h a1 = ldA(A + (size_t)(16 + lh) * lda + k + lq * 16);
    v16h b0 = ldB(Bw + (size_t)lh        * ldb + k + lq * 16);
    v16h b1 = ldB(Bw + (size_t)(16 + lh) * ldb + k + lq * 16);
    wmma4(a0, a1, b0, b1, c);
  }
}

// C 16x16 f32 layout: lane n = lane&15, VGPR j -> m = j + (lane>>4)*8.
#define WAVE_TILE_COORDS()                                   \
  const int wave  = threadIdx.x >> 5;                        \
  const int lane  = threadIdx.x & 31;                        \
  const int lh    = lane & 15;                               \
  const int lq    = lane >> 4;                               \
  const int m0    = blockIdx.x * 128 + (wave & 3) * 32;      \
  const int n0    = blockIdx.y * 64  + (wave >> 2) * 32;     \
  (void)lh; (void)lq; (void)m0; (void)n0;

// ---- TDM descriptor + issue (ISA ch.8) -------------------------------------
#if USE_TDM
// Load a [rows x 64]-f16 2D tile starting at g (row stride = strideElems f16)
// into LDS at byte offset ldsOff, padding each 128B row to 144B.
__device__ __forceinline__ void tdm_load_tile(const _Float16* g, int strideElems,
                                              int rows, unsigned ldsOff) {
  unsigned long long ga = (unsigned long long)g;
  unsigned td0 = (unsigned)strideElems;   // tensor dim0 (>= tile_dim0)
  unsigned td1 = 0x100000u;               // tensor dim1 (>= rows)
  unsigned long long s0 = (unsigned long long)(unsigned)strideElems;
  u32x4 g0;
  g0[0] = 1u;                                                  // count=1, user desc
  g0[1] = ldsOff;                                              // lds_addr
  g0[2] = (unsigned)(ga & 0xffffffffull);                      // global_addr[31:0]
  g0[3] = (unsigned)((ga >> 32) & 0x01ffffffull) | 0x80000000u;// addr[56:32] | type=2
  i32x8 g1;
  g1[0] = (int)((1u << 16)      // data_size = 1 -> 2 bytes
              | (1u << 20)      // pad_enable
              | (4u << 22)      // pad_interval: 32 DWORDs (one 128B row)
              | (3u << 25));    // pad_amount:   4 DWORDs (16B)
  g1[1] = (int)((td0 & 0xffffu) << 16);                        // [63:48]=td0 lo
  g1[2] = (int)(((td0 >> 16) & 0xffffu) | ((td1 & 0xffffu) << 16));
  g1[3] = (int)(((td1 >> 16) & 0xffffu) | ((unsigned)KSTEP << 16)); // tile_dim0 = 64
  g1[4] = (int)((unsigned)rows & 0xffffu);                     // tile_dim1, tile_dim2=0
  g1[5] = (int)(unsigned)(s0 & 0xffffffffull);                 // dim0_stride lo
  g1[6] = (int)(unsigned)((s0 >> 32) & 0xffffull);             // dim0_stride hi
  g1[7] = 0;                                                   // dim1_stride (2D)
  i32x4 z4 = {0, 0, 0, 0};
#if defined(__clang_major__) && __clang_major__ >= 23
  i32x8 z8 = {0, 0, 0, 0, 0, 0, 0, 0};
  __builtin_amdgcn_tensor_load_to_lds(g0, g1, z4, z4, z8, 0);
#else
  __builtin_amdgcn_tensor_load_to_lds(g0, g1, z4, z4, 0);
#endif
}
#endif

// ---------------------------------------------------------------------------
__global__ void k_zero_u32(unsigned* p) { p[0] = 0u; }

__global__ void k_f32_to_f16(const float* __restrict__ x, _Float16* __restrict__ y, int n) {
  int i = blockIdx.x * blockDim.x + threadIdx.x;
  if (i < n) y[i] = (_Float16)x[i];
}

// ---- LayerNorm: one row per block (256 threads) ----------------------------
__device__ __forceinline__ float block_red(float v, float* red, int isMax) {
  int tid = threadIdx.x;
  red[tid] = v;
  __syncthreads();
  for (int s = 128; s > 0; s >>= 1) {
    if (tid < s) red[tid] = isMax ? fmaxf(red[tid], red[tid + s]) : (red[tid] + red[tid + s]);
    __syncthreads();
  }
  float r = red[0];
  __syncthreads();
  return r;
}

__global__ void k_layernorm(const float* __restrict__ X, const float* __restrict__ g,
                            const float* __restrict__ bta, float* __restrict__ Y32,
                            _Float16* __restrict__ Y16, int cols) {
  __shared__ float red[256];
  const size_t row = blockIdx.x;
  const float* x = X + row * cols;
  float s = 0.f;
  for (int c = threadIdx.x; c < cols; c += 256) s += x[c];
  float mu = block_red(s, red, 0) / (float)cols;
  float sq = 0.f;
  for (int c = threadIdx.x; c < cols; c += 256) { float d = x[c] - mu; sq += d * d; }
  float var = block_red(sq, red, 0) / (float)cols;
  float rstd = rsqrtf(var + EPS);
  for (int c = threadIdx.x; c < cols; c += 256) {
    float y = (x[c] - mu) * rstd * g[c] + bta[c];
    Y32[row * cols + c] = y;
    Y16[row * cols + c] = (_Float16)y;
  }
}

// ---- TDM-staged GEMM: C = epilogue(A[M,K] @ W[N,K]^T) ----------------------
// mode 0: f16 C[m*ldc+n] = (acc+bias)*scale
// mode 1: f16 per-head transposed [B,H,Dh,T] (for attention B-operands)
// mode 2: f32 Out[m*ldc+n] = resid + gamma*(acc+bias)
__global__ __launch_bounds__(256) void k_gemm_staged(
    const _Float16* __restrict__ A, int lda,
    const _Float16* __restrict__ W, int ldb,
    const float* __restrict__ bias, const float* __restrict__ gamma,
    const float* __restrict__ resid, void* __restrict__ Cout,
    int ldc, int K, float scale, int mode, int T) {
  __shared__ _Float16 smem[2][A_HALVES + B_HALVES];   // 54 KB
  WAVE_TILE_COORDS();
  const int m0blk = blockIdx.x * 128;
  const int n0blk = blockIdx.y * 64;
  const _Float16* Ag = A + (size_t)m0blk * lda;
  const _Float16* Bg = W + (size_t)n0blk * ldb;
  const int nstages = K / KSTEP;

#if USE_TDM
  if (threadIdx.x < 32) {
    unsigned base = (unsigned)(size_t)(&smem[0][0]);
    tdm_load_tile(Ag, lda, 128, base);
    tdm_load_tile(Bg, ldb, 64, base + A_HALVES * 2);
  }
#else
  {
    for (int c = threadIdx.x; c < 1024; c += 256) {
      int r = c >> 3, q = c & 7;
      *(v8h*)&smem[0][r * LROW + q * 8] = *(const v8h*)(Ag + (size_t)r * lda + q * 8);
    }
    for (int c = threadIdx.x; c < 512; c += 256) {
      int r = c >> 3, q = c & 7;
      *(v8h*)&smem[0][A_HALVES + r * LROW + q * 8] = *(const v8h*)(Bg + (size_t)r * ldb + q * 8);
    }
  }
#endif

  v8f c[2][2] = {};
  for (int s = 0; s < nstages; ++s) {
    const int buf = s & 1;
#if USE_TDM
    if (threadIdx.x < 32) {
      if (s + 1 < nstages) {
        unsigned nb = (unsigned)(size_t)(&smem[buf ^ 1][0]);
        tdm_load_tile(Ag + (s + 1) * KSTEP, lda, 128, nb);
        tdm_load_tile(Bg + (s + 1) * KSTEP, ldb, 64, nb + A_HALVES * 2);
        __builtin_amdgcn_s_wait_tensorcnt(2);   // stage s landed
      } else {
        __builtin_amdgcn_s_wait_tensorcnt(0);
      }
    }
#else
    if (s + 1 < nstages) {
      const int bn = buf ^ 1;
      for (int cc = threadIdx.x; cc < 1024; cc += 256) {
        int r = cc >> 3, q = cc & 7;
        *(v8h*)&smem[bn][r * LROW + q * 8] =
            *(const v8h*)(Ag + (size_t)r * lda + (s + 1) * KSTEP + q * 8);
      }
      for (int cc = threadIdx.x; cc < 512; cc += 256) {
        int r = cc >> 3, q = cc & 7;
        *(v8h*)&smem[bn][A_HALVES + r * LROW + q * 8] =
            *(const v8h*)(Bg + (size_t)r * ldb + (s + 1) * KSTEP + q * 8);
      }
    }
#endif
    __syncthreads();   // stage s visible to all waves
    const _Float16* As = &smem[buf][0]        + (wave & 3) * 32 * LROW;
    const _Float16* Bs = &smem[buf][A_HALVES] + (wave >> 2) * 32 * LROW;
#pragma unroll
    for (int ks = 0; ks < KSTEP; ks += 32) {
      v16h a0 = ldA(As + lh * LROW + ks + lq * 16);
      v16h a1 = ldA(As + (16 + lh) * LROW + ks + lq * 16);
      v16h b0 = ldB(Bs + lh * LROW + ks + lq * 16);
      v16h b1 = ldB(Bs + (16 + lh) * LROW + ks + lq * 16);
      wmma4(a0, a1, b0, b1, c);
    }
    __syncthreads();   // done reading buf before it is overwritten
  }

#pragma unroll
  for (int i = 0; i < 2; ++i)
#pragma unroll
    for (int j = 0; j < 2; ++j)
#pragma unroll
      for (int e = 0; e < 8; ++e) {
        int gm = m0 + i * 16 + lq * 8 + e;
        int gn = n0 + j * 16 + lh;
        float acc = c[i][j][e];
        if (mode == 2) {
          size_t idx = (size_t)gm * ldc + gn;
          ((float*)Cout)[idx] = resid[idx] + gamma[gn] * (acc + bias[gn]);
        } else {
          float v = (acc + bias[gn]) * scale;
          size_t idx;
          if (mode == 1) {
            int bb = gm / T, tt = gm - bb * T;
            int hh = gn >> 6, dd = gn & 63;
            idx = (((size_t)bb * H + hh) * Dh + dd) * T + tt;
          } else {
            idx = (size_t)gm * ldc + gn;
          }
          ((_Float16*)Cout)[idx] = (_Float16)v;
        }
      }
}

// ---- scores: S[b,h,tv,tl] = q . k, plus running global max -----------------
__global__ void k_scores(const _Float16* __restrict__ Q, const _Float16* __restrict__ Kc,
                         float* __restrict__ S, unsigned* __restrict__ gmax) {
  WAVE_TILE_COORDS();
  const int z = blockIdx.z;              // b*H + h
  const int bb = z / H, hh = z - bb * H;
  const _Float16* Ab = Q  + ((size_t)bb * TV) * E + hh * Dh;
  const _Float16* Bb = Kc + ((size_t)bb * TL) * E + hh * Dh;
  v8f c[2][2] = {};
  mm32x32(Ab + (size_t)m0 * E, E, Bb + (size_t)n0 * E, E, Dh, c);
  float lmax = -INFINITY;
  float* Sb = S + (size_t)z * TV * TL;
#pragma unroll
  for (int i = 0; i < 2; ++i)
#pragma unroll
    for (int j = 0; j < 2; ++j)
#pragma unroll
      for (int e = 0; e < 8; ++e) {
        int gm = m0 + i * 16 + lq * 8 + e;
        int gn = n0 + j * 16 + lh;
        float v = c[i][j][e];
        Sb[(size_t)gm * TL + gn] = v;
        lmax = fmaxf(lmax, v);
      }
  for (int off = 16; off > 0; off >>= 1)
    lmax = fmaxf(lmax, __shfl_xor(lmax, off, 32));
  if (lane == 0) atomicMax(gmax, enc_f32(lmax));
}

// ---- softmax v->l: rows of length TL=256, mask over text tokens ------------
__global__ void k_softmax_v(const float* __restrict__ S, const unsigned* __restrict__ gmax,
                            const unsigned char* __restrict__ mask_l,
                            _Float16* __restrict__ A) {
  __shared__ float red[256];
  const size_t r = blockIdx.x;                 // (b*H + h)*TV + tv
  const int bb = (int)(r / ((size_t)H * TV));
  const int t = threadIdx.x;
  const float gm = dec_f32(gmax[0]);
  float v = S[r * TL + t] - gm;
  v = fminf(fmaxf(v, -CLAMP), CLAMP);
  if (mask_l[bb * TL + t]) v = -INFINITY;
  float m = block_red(v, red, 1);
  float e = __expf(v - m);                     // exp(-inf)=0 for masked
  float s = block_red(e, red, 0);
  A[r * TL + t] = (_Float16)(e / s);
}

// ---- softmax l->v: rows of length TV=4096 (column reads), mask over vision -
__global__ void k_softmax_l(const float* __restrict__ S, const unsigned* __restrict__ gmax,
                            const unsigned char* __restrict__ mask_v,
                            _Float16* __restrict__ A) {
  __shared__ float sbuf[TV];
  __shared__ float red[256];
  const int r  = blockIdx.x;                   // (b*H + h)*TL + tl
  const int bb = r / (H * TL);
  const int hh = (r / TL) % H;
  const int tl = r % TL;
  const float gm = dec_f32(gmax[0]);
  const float* Sb = S + (((size_t)bb * H + hh) * TV) * TL + tl;

  float lm = -INFINITY;
  for (int it = 0; it < TV / 256; ++it) {
    int tv = it * 256 + threadIdx.x;
    float v = Sb[(size_t)tv * TL] - gm;
    v = fminf(fmaxf(v, -CLAMP), CLAMP);
    sbuf[tv] = v;
    lm = fmaxf(lm, v);
  }
  float m1 = block_red(lm, red, 1);            // row-max BEFORE masking (ref order)

  float lm2 = -INFINITY;
  for (int it = 0; it < TV / 256; ++it) {
    int tv = it * 256 + threadIdx.x;
    float v = sbuf[tv] - m1;
    v = fminf(fmaxf(v, -CLAMP), CLAMP);
    if (mask_v[bb * TV + tv]) v = -INFINITY;
    sbuf[tv] = v;
    lm2 = fmaxf(lm2, v);
  }
  __syncthreads();
  float m2 = block_red(lm2, red, 1);

  float ls = 0.f;
  for (int it = 0; it < TV / 256; ++it) {
    int tv = it * 256 + threadIdx.x;
    float e = __expf(sbuf[tv] - m2);
    sbuf[tv] = e;
    ls += e;
  }
  __syncthreads();
  float sum = block_red(ls, red, 0);

  _Float16* Ab = A + ((((size_t)bb * H + hh) * TL) + tl) * TV;
  for (int it = 0; it < TV / 256; ++it) {
    int tv = it * 256 + threadIdx.x;
    Ab[tv] = (_Float16)(sbuf[tv] / sum);
  }
}

// ---- attention output GEMM (batched over b*H): C[T,Dh] = attn @ valT^T -----
__global__ void k_attn_out(const _Float16* __restrict__ A, long aStride, int lda,
                           const _Float16* __restrict__ Bw, long bStride, int ldb,
                           _Float16* __restrict__ C, int T, int ldc, int K) {
  WAVE_TILE_COORDS();
  const int z = blockIdx.z;
  const int bb = z / H, hh = z - bb * H;
  const _Float16* Ab = A + (size_t)z * aStride;
  const _Float16* Bb = Bw + (size_t)z * bStride;
  v8f c[2][2] = {};
  mm32x32(Ab + (size_t)m0 * lda, lda, Bb + (size_t)n0 * ldb, ldb, K, c);
  _Float16* Cb = C + (size_t)bb * T * ldc + hh * Dh;
#pragma unroll
  for (int i = 0; i < 2; ++i)
#pragma unroll
    for (int j = 0; j < 2; ++j)
#pragma unroll
      for (int e = 0; e < 8; ++e) {
        int gm = m0 + i * 16 + lq * 8 + e;
        int gn = n0 + j * 16 + lh;
        Cb[(size_t)gm * ldc + gn] = (_Float16)c[i][j][e];
      }
}

// ---------------------------------------------------------------------------
extern "C" void kernel_launch(void* const* d_in, const int* in_sizes, int n_in,
                              void* d_out, int out_size, void* d_ws, size_t ws_size,
                              hipStream_t stream) {
  (void)in_sizes; (void)n_in; (void)out_size; (void)ws_size;
  const float* v      = (const float*)d_in[0];
  const float* l      = (const float*)d_in[1];
  const unsigned char* mask_v = (const unsigned char*)d_in[2];
  const unsigned char* mask_l = (const unsigned char*)d_in[3];
  const float* ln_v_g = (const float*)d_in[4];
  const float* ln_v_b = (const float*)d_in[5];
  const float* ln_l_g = (const float*)d_in[6];
  const float* ln_l_b = (const float*)d_in[7];
  const float* Wq  = (const float*)d_in[8];   const float* bq  = (const float*)d_in[9];
  const float* Wk  = (const float*)d_in[10];  const float* bk  = (const float*)d_in[11];
  const float* Wvv = (const float*)d_in[12];  const float* bvv = (const float*)d_in[13];
  const float* Wvl = (const float*)d_in[14];  const float* bvl = (const float*)d_in[15];
  const float* Wov = (const float*)d_in[16];  const float* bov = (const float*)d_in[17];
  const float* Wol = (const float*)d_in[18];  const float* bol = (const float*)d_in[19];
  const float* gamma_v = (const float*)d_in[20];
  const float* gamma_l = (const float*)d_in[21];

  float* out_v = (float*)d_out;                       // [B,TV,VD]
  float* out_l = out_v + (size_t)B * TV * VD;         // [B,TL,LD]

  // ---- workspace carve-up (256B aligned) ----
  char* base = (char*)d_ws;
  size_t cur = 0;
  auto alloc = [&](size_t bytes) -> void* {
    void* p = base + cur;
    cur = (cur + bytes + 255) & ~(size_t)255;
    return p;
  };
  float*    vn32   = (float*)   alloc((size_t)B * TV * VD * 4);
  float*    ln32   = (float*)   alloc((size_t)B * TL * LD * 4);
  _Float16* vn16   = (_Float16*)alloc((size_t)B * TV * VD * 2);
  _Float16* ln16   = (_Float16*)alloc((size_t)B * TL * LD * 2);
  _Float16* wq16   = (_Float16*)alloc((size_t)E * VD * 2);
  _Float16* wk16   = (_Float16*)alloc((size_t)E * LD * 2);
  _Float16* wvv16  = (_Float16*)alloc((size_t)E * VD * 2);
  _Float16* wvl16  = (_Float16*)alloc((size_t)E * LD * 2);
  _Float16* wov16  = (_Float16*)alloc((size_t)VD * E * 2);
  _Float16* wol16  = (_Float16*)alloc((size_t)LD * E * 2);
  _Float16* q16    = (_Float16*)alloc((size_t)B * TV * E * 2);
  _Float16* k16    = (_Float16*)alloc((size_t)B * TL * E * 2);
  _Float16* vvT16  = (_Float16*)alloc((size_t)B * H * Dh * TV * 2);
  _Float16* vlT16  = (_Float16*)alloc((size_t)B * H * Dh * TL * 2);
  float*    sco32  = (float*)   alloc((size_t)B * H * TV * TL * 4);
  _Float16* attV16 = (_Float16*)alloc((size_t)B * H * TV * TL * 2);
  _Float16* attL16 = (_Float16*)alloc((size_t)B * H * TL * TV * 2);
  _Float16* ov16   = (_Float16*)alloc((size_t)B * TV * E * 2);
  _Float16* ol16   = (_Float16*)alloc((size_t)B * TL * E * 2);
  unsigned* gmax   = (unsigned*)alloc(256);

  k_zero_u32<<<1, 1, 0, stream>>>(gmax);

  auto cast16 = [&](const float* src, _Float16* dst, int n) {
    k_f32_to_f16<<<(n + 255) / 256, 256, 0, stream>>>(src, dst, n);
  };
  cast16(Wq,  wq16,  E * VD);
  cast16(Wk,  wk16,  E * LD);
  cast16(Wvv, wvv16, E * VD);
  cast16(Wvl, wvl16, E * LD);
  cast16(Wov, wov16, VD * E);
  cast16(Wol, wol16, LD * E);

  k_layernorm<<<B * TV, 256, 0, stream>>>(v, ln_v_g, ln_v_b, vn32, vn16, VD);
  k_layernorm<<<B * TL, 256, 0, stream>>>(l, ln_l_g, ln_l_b, ln32, ln16, LD);

  // projections (TDM-staged): q (scaled), k, per-head-transposed values
  k_gemm_staged<<<dim3(B * TV / 128, E / 64), 256, 0, stream>>>(
      vn16, VD, wq16, VD, bq, nullptr, nullptr, q16, E, VD, SCALE, 0, TV);
  k_gemm_staged<<<dim3(B * TL / 128, E / 64), 256, 0, stream>>>(
      ln16, LD, wk16, LD, bk, nullptr, nullptr, k16, E, LD, 1.0f, 0, TL);
  k_gemm_staged<<<dim3(B * TV / 128, E / 64), 256, 0, stream>>>(
      vn16, VD, wvv16, VD, bvv, nullptr, nullptr, vvT16, E, VD, 1.0f, 1, TV);
  k_gemm_staged<<<dim3(B * TL / 128, E / 64), 256, 0, stream>>>(
      ln16, LD, wvl16, LD, bvl, nullptr, nullptr, vlT16, E, LD, 1.0f, 1, TL);

  // scores + global max
  k_scores<<<dim3(TV / 128, TL / 64, B * H), 256, 0, stream>>>(q16, k16, sco32, gmax);

  // dual softmax
  k_softmax_v<<<B * H * TV, 256, 0, stream>>>(sco32, gmax, mask_l, attV16);
  k_softmax_l<<<B * H * TL, 256, 0, stream>>>(sco32, gmax, mask_v, attL16);

  // attention outputs
  k_attn_out<<<dim3(TV / 128, 1, B * H), 256, 0, stream>>>(
      attV16, (long)TV * TL, TL, vlT16, (long)Dh * TL, TL, ov16, TV, E, TL);
  k_attn_out<<<dim3(TL / 128, 1, B * H), 256, 0, stream>>>(
      attL16, (long)TL * TV, TV, vvT16, (long)Dh * TV, TV, ol16, TL, E, TV);

  // output projections + scaled residual (TDM-staged)
  k_gemm_staged<<<dim3(B * TV / 128, VD / 64), 256, 0, stream>>>(
      ov16, E, wov16, E, bov, gamma_v, vn32, out_v, VD, E, 1.0f, 2, TV);
  k_gemm_staged<<<dim3(B * TL / 128, LD / 64), 256, 0, stream>>>(
      ol16, E, wol16, E, bol, gamma_l, ln32, out_l, LD, E, 1.0f, 2, TL);
}
